// GCIQEValue_3934190043553
// MI455X (gfx1250) — matrix operand order repaired and checked by reading, not compiled, not run on hardware
//
#include <hip/hip_runtime.h>
#include <math.h>

// ---------- types matching CDNA5 WMMA operand layouts ----------
typedef __attribute__((ext_vector_type(16))) __bf16 bf16x16;
typedef __attribute__((ext_vector_type(8)))  float  f32x8;

struct U4 { unsigned x, y, z, w; };
struct U2 { unsigned x, y; };
union BF16Frag { bf16x16 v; U4 q[2]; };

static __device__ inline unsigned f2bfbits(float x) {
    union { float f; unsigned u; } v; v.f = x;
    return (v.u + 0x7FFFu + ((v.u >> 16) & 1u)) >> 16;   // RNE bf16 bits
}
static __device__ inline unsigned pack2bf(float a, float b) {
    return f2bfbits(a) | (f2bfbits(b) << 16);
}

// Branch-free gelu (tanh form) on native v_exp_f32 / v_rcp_f32.
static __device__ inline float gelu_f(float x) {
    const float k = 0.7978845608028654f;                 // sqrt(2/pi)
    float u = k * (x + 0.044715f * x * x * x);
    float e = __expf(2.0f * u);
    float th = 1.0f - 2.0f * __builtin_amdgcn_rcpf(e + 1.0f);
    return 0.5f * x * (1.0f + th);
}

// ---------- weight prep: W[K,N] f32 (k-major) -> Wt[N,K] bf16 bits ----------
__global__ void transpose_bf16(const float* __restrict__ W,
                               unsigned short* __restrict__ Wt, int K, int N) {
    int i = blockIdx.x * blockDim.x + threadIdx.x;
    if (i >= K * N) return;
    int n = i / K, k = i - n * K;
    Wt[i] = (unsigned short)f2bfbits(W[(size_t)k * N + n]);
}

// ---------- WMMA fragment loaders ----------
// A 16x32 bf16 from LDS (row-major, stride = K elements).
static __device__ inline bf16x16 load_a_lds(const unsigned short* base, int stride,
                                            int k0, int lane) {
    const int m = lane & 15;
    const int hiofs = (lane >> 4) << 3;
    const unsigned short* p = base + m * stride + k0 + hiofs;
    BF16Frag f;
    f.q[0] = *(const U4*)(p);                            // ds_load_b128
    f.q[1] = *(const U4*)(p + 16);
    return f.v;
}

// B 32x16 bf16 from global Wt[N,K] (n-major).
static __device__ inline bf16x16 load_b_glb(const unsigned short* __restrict__ Wt,
                                            int ldk, int n0, int k0, int lane) {
    const int n  = n0 + (lane & 15);
    const int kb = k0 + ((lane >> 4) << 3);
    const unsigned short* p = Wt + (size_t)n * ldk + kb;
    BF16Frag f;
    f.q[0] = *(const U4*)(p);                            // global_load_b128
    f.q[1] = *(const U4*)(p + 16);
    return f.v;
}

// One wave: 16-row x 512-col layer; A resident in registers, weights streamed.
template <int K, bool GELU>
static __device__ inline void gemm16(const unsigned short* aBase,
                                     const unsigned short* __restrict__ Wt,
                                     const float* __restrict__ bias,
                                     float* outBase) {
    const int lane = threadIdx.x & 31;
    const int hi = lane >> 4;
    const int nn = lane & 15;
    constexpr int KT = K / 32;

    bf16x16 afrag[KT];
#pragma unroll
    for (int kt = 0; kt < KT; ++kt)
        afrag[kt] = load_a_lds(aBase, K, kt * 32, lane);

    __builtin_prefetch(Wt, 0, 3);                        // global_prefetch_b8 (near)
#pragma unroll 1
    for (int nt = 0; nt < 32; ++nt) {
        const int n0 = nt << 4;
        f32x8 acc = {0.f, 0.f, 0.f, 0.f, 0.f, 0.f, 0.f, 0.f};
#pragma unroll
        for (int kt = 0; kt < KT; ++kt) {
            bf16x16 bv = load_b_glb(Wt, K, n0, kt * 32, lane);
            acc = __builtin_amdgcn_wmma_f32_16x16x32_bf16(
                false, afrag[kt], false, bv, (short)0, acc, false, false);
        }
        const float bn = bias[n0 + nn];
#pragma unroll
        for (int r = 0; r < 8; ++r) {                    // D: M=r+8*hi, N=lane&15
            float v = acc[r] + bn;
            if (GELU) v = gelu_f(v);
            outBase[(r + (hi << 3)) * 512 + n0 + nn] = v;
        }
    }
}

// 64 threads, 32 rows: 2 lanes/row, 256 cols each; float4 reads, packed bf16 writes.
static __device__ inline void layernorm_block(const float* sAct, unsigned short* sIn,
                                              const float* __restrict__ g,
                                              const float* __restrict__ b) {
    const int t = threadIdx.x;
    const int row = t >> 1, half = t & 1;
    const float* p = sAct + row * 512 + half * 256;
    float s = 0.f, s2 = 0.f;
#pragma unroll 4
    for (int i = 0; i < 256; i += 4) {
        float4 v = *(const float4*)(p + i);
        s  += (v.x + v.y) + (v.z + v.w);
        s2 += (v.x * v.x + v.y * v.y) + (v.z * v.z + v.w * v.w);
    }
    s  += __shfl_xor(s, 1, 32);
    s2 += __shfl_xor(s2, 1, 32);
    const float mean = s * (1.0f / 512.0f);
    const float var  = s2 * (1.0f / 512.0f) - mean * mean;
    const float inv  = rsqrtf(var + 1e-6f);
    unsigned short* q = sIn + row * 512 + half * 256;
    const float* gg = g + half * 256;
    const float* bb = b + half * 256;
#pragma unroll 4
    for (int i = 0; i < 256; i += 4) {
        float4 v = *(const float4*)(p + i);
        float4 gv = *(const float4*)(gg + i);
        float4 bv = *(const float4*)(bb + i);
        U2 o;
        o.x = pack2bf((v.x - mean) * inv * gv.x + bv.x,
                      (v.y - mean) * inv * gv.y + bv.y);
        o.y = pack2bf((v.z - mean) * inv * gv.z + bv.z,
                      (v.w - mean) * inv * gv.w + bv.w);
        *(U2*)(q + i) = o;                               // ds_store_b64
    }
}

// IQE component == measure of union of intervals [x_j, y_j) over dims with x_j < y_j.
// (The reference's sort/cumsum/neg_f construction telescopes to exactly this; ties
// contribute zero length, so tie-order is irrelevant.) O(32^2), branch-free, no scratch.
static __device__ inline float iqe_component(const float* x, const float* y) {
    float comp = 0.f;
#pragma unroll 1
    for (int j = 0; j < 32; ++j) {
        const float xj = x[j], yj = y[j];
        float E = -3.4e38f;                              // max y over intervals before j
#pragma unroll 8
        for (int k = 0; k < 32; ++k) {
            const float xk = x[k], yk = y[k];
            const bool before = (xk < xj) || (xk == xj && k < j);
            const bool use = before && (xk < yk);        // only valid intervals
            E = use ? fmaxf(E, yk) : E;
        }
        const float c = fmaxf(yj - fmaxf(xj, E), 0.f);
        comp += (xj < yj) ? c : 0.f;
    }
    return comp;
}

// ---------- fused phi(obs), phi(goals), IQE ----------
__global__ __launch_bounds__(64) void iqe_fused(
    const float* __restrict__ obs, const float* __restrict__ goals,
    const unsigned short* __restrict__ wt,               // Wt0|Wt1|Wt2|Wt3 bf16 bits
    const float* __restrict__ b0, const float* __restrict__ ln0s, const float* __restrict__ ln0b,
    const float* __restrict__ b1, const float* __restrict__ ln1s, const float* __restrict__ ln1b,
    const float* __restrict__ b2, const float* __restrict__ ln2s, const float* __restrict__ ln2b,
    const float* __restrict__ b3, const float* __restrict__ alpha,
    float* __restrict__ out) {
    extern __shared__ char smem[];
    unsigned short* sIn = (unsigned short*)smem;                   // 32 KB
    float* sAct  = (float*)(smem + 32 * 512 * 2);                  // 64 KB
    float* sPhiS = (float*)(smem + 32 * 512 * 2 + 32 * 512 * 4);   // 64 KB

    const int t = threadIdx.x;
    const int wave = t >> 5;
    const size_t row0 = (size_t)blockIdx.x * 32;

    const unsigned short* Wt0 = wt;
    const unsigned short* Wt1 = wt + 32768;
    const unsigned short* Wt2 = wt + 32768 + 262144;
    const unsigned short* Wt3 = wt + 32768 + 524288;

#pragma unroll 1
    for (int pass = 0; pass < 2; ++pass) {
        const float* src = pass ? goals : obs;
        float* phiOut = pass ? sAct : sPhiS;

        {   // stage 32 rows (64 f32 each) -> bf16 LDS (stride 64), float4 + packed stores
            const int r = t >> 1, c0 = (t & 1) * 32;
            const float* p = src + (row0 + r) * 64 + c0;
            unsigned short* q = sIn + r * 64 + c0;
#pragma unroll
            for (int i = 0; i < 32; i += 4) {
                float4 v = *(const float4*)(p + i);
                U2 o; o.x = pack2bf(v.x, v.y); o.y = pack2bf(v.z, v.w);
                *(U2*)(q + i) = o;
            }
        }
        __syncthreads();

        const unsigned short* aIn64  = sIn + wave * 16 * 64;
        const unsigned short* aIn512 = sIn + wave * 16 * 512;
        float* myAct = sAct + wave * 16 * 512;

        gemm16<64, true>(aIn64, Wt0, b0, myAct);
        __syncthreads();
        layernorm_block(sAct, sIn, ln0s, ln0b);
        __syncthreads();

        gemm16<512, true>(aIn512, Wt1, b1, myAct);
        __syncthreads();
        layernorm_block(sAct, sIn, ln1s, ln1b);
        __syncthreads();

        gemm16<512, true>(aIn512, Wt2, b2, myAct);
        __syncthreads();
        layernorm_block(sAct, sIn, ln2s, ln2b);
        __syncthreads();

        gemm16<512, false>(aIn512, Wt3, b3, phiOut + wave * 16 * 512); // phi
        __syncthreads();
    }

    // ---- IQE: phi_s in sPhiS, phi_g in sAct; 2 threads/row x 8 components ----
    const float a = __builtin_amdgcn_rcpf(1.0f + __expf(-alpha[0]));
    const int row = t >> 1;
    const int cbase = (t & 1) * 8;
    const float* xr = sPhiS + row * 512;
    const float* yr = sAct + row * 512;
    float ssum = 0.f, smax = -3.4e38f;
#pragma unroll 1
    for (int c = 0; c < 8; ++c) {
        const float comp = iqe_component(xr + (cbase + c) * 32, yr + (cbase + c) * 32);
        ssum += comp;
        smax = fmaxf(smax, comp);
    }
    ssum += __shfl_xor(ssum, 1, 32);
    smax = fmaxf(smax, __shfl_xor(smax, 1, 32));
    if ((t & 1) == 0)
        out[row0 + row] = a * (ssum * (1.0f / 16.0f)) + (1.0f - a) * smax;
}

extern "C" void kernel_launch(void* const* d_in, const int* in_sizes, int n_in,
                              void* d_out, int out_size, void* d_ws, size_t ws_size,
                              hipStream_t stream) {
    const float* obs   = (const float*)d_in[0];
    const float* goals = (const float*)d_in[1];
    const float* W0 = (const float*)d_in[2];
    const float* b0 = (const float*)d_in[3];
    const float* ln0s = (const float*)d_in[4];
    const float* ln0b = (const float*)d_in[5];
    const float* W1 = (const float*)d_in[6];
    const float* b1 = (const float*)d_in[7];
    const float* ln1s = (const float*)d_in[8];
    const float* ln1b = (const float*)d_in[9];
    const float* W2 = (const float*)d_in[10];
    const float* b2 = (const float*)d_in[11];
    const float* ln2s = (const float*)d_in[12];
    const float* ln2b = (const float*)d_in[13];
    const float* W3 = (const float*)d_in[14];
    const float* b3 = (const float*)d_in[15];
    const float* alpha = (const float*)d_in[16];
    unsigned short* wt = (unsigned short*)d_ws;

    transpose_bf16<<<(32768 + 255) / 256, 256, 0, stream>>>(W0, wt, 64, 512);
    transpose_bf16<<<(262144 + 255) / 256, 256, 0, stream>>>(W1, wt + 32768, 512, 512);
    transpose_bf16<<<(262144 + 255) / 256, 256, 0, stream>>>(W2, wt + 32768 + 262144, 512, 512);
    transpose_bf16<<<(262144 + 255) / 256, 256, 0, stream>>>(W3, wt + 32768 + 524288, 512, 512);

    const int rows_per_block = 32;
    dim3 grid(131072 / rows_per_block);
    const size_t shmem = 32 * 512 * 2 + 32 * 512 * 4 + 32 * 512 * 4;  // 160 KB dynamic LDS
    iqe_fused<<<grid, 64, shmem, stream>>>(obs, goals, wt,
                                           b0, ln0s, ln0b, b1, ln1s, ln1b,
                                           b2, ln2s, ln2b, b3, alpha,
                                           (float*)d_out);
}
